// Generator_59184649339010
// MI455X (gfx1250) — matrix operand order repaired and checked by reading, not compiled, not run on hardware
//
#include <hip/hip_runtime.h>
#include <stdint.h>

// Problem constants
#define B_ 512
#define T_ 128
#define D_ 64
#define H_ 512
#define C_ 10
#define N4H 2048           // 4*H
#define MISSING_V 128.0f
#define NPRED 4161536LL    // B*(T-1)*D

typedef __attribute__((ext_vector_type(16))) __bf16 v16bf;
typedef __attribute__((ext_vector_type(8)))  float  v8f;
typedef __attribute__((ext_vector_type(4)))  unsigned int v4u;
typedef __attribute__((ext_vector_type(4)))  unsigned int u32x4;
typedef __attribute__((ext_vector_type(8)))  int i32x8;
typedef __attribute__((ext_vector_type(4)))  int i32x4;
typedef unsigned short u16;

#if __has_builtin(__builtin_amdgcn_tensor_load_to_lds) && \
    __has_builtin(__builtin_amdgcn_s_wait_tensorcnt)
#define HAVE_TDM 1
#else
#define HAVE_TDM 0
#endif

// ---------- helpers ----------
__device__ __forceinline__ u16 f2bf(float f) {
  unsigned u = __float_as_uint(f);
  u += 0x7fffu + ((u >> 16) & 1u);       // round-to-nearest-even
  return (u16)(u >> 16);
}

// A-fragment (16x32 bf16): lane holds row (lane&15); elems 0..15 = K base..base+15,
// base = 16*(lane>>4).
__device__ __forceinline__ v16bf loadA(const u16* p) {
  union { v4u u[2]; v16bf v; } t;
  t.u[0] = *(const v4u*)(p);
  t.u[1] = *(const v4u*)(p + 8);
  return t.v;
}
// B-fragment (32x16 bf16) from col-major-packed rows: lane holds col (lane&15),
// 16 contiguous K values starting at 16*(lane>>4).
__device__ __forceinline__ v16bf loadB(const u16* p) {
  union { v4u u[2]; v16bf v; } t;
  t.u[0] = *(const v4u*)(p);
  t.u[1] = *(const v4u*)(p + 8);
  return t.v;
}

__device__ __forceinline__ v8f wmma_bf16(v16bf a, v16bf b, v8f c) {
  return __builtin_amdgcn_wmma_f32_16x16x32_bf16(false, a, false, b,
                                                 (short)0, c, false, false);
}

__device__ __forceinline__ float sigm(float x) { return 1.0f / (1.0f + expf(-x)); }

#if HAVE_TDM
// Issue one TDM 2-D tile load: 64 rows of WT[.][ldw] starting at rowStart,
// K-columns [kStart, kStart+tile0), into LDS byte offset ldsOff.
__device__ __forceinline__ void tdm_load_strip(const u16* WT, int ldw, int rowStart,
                                               int kStart, int tile0, unsigned ldsOff) {
  unsigned long long gaddr =
      (unsigned long long)(const char*)(WT + (size_t)rowStart * ldw + kStart);
  u32x4 g0;
  g0[0] = 1u;                                           // count=1 valid descriptor
  g0[1] = ldsOff;                                       // lds_addr (bytes)
  g0[2] = (unsigned)(gaddr & 0xffffffffu);              // global_addr lo
  g0[3] = (unsigned)((gaddr >> 32) & 0x01ffffffu) | 0x80000000u;  // addr hi | type=2
  i32x8 g1;
  g1[0] = (int)(1u << 16);                              // data_size = 2 bytes
  g1[1] = (int)((unsigned)(ldw & 0xffff) << 16);        // tensor_dim0 lo16
  g1[2] = (int)(((unsigned)ldw >> 16) | (64u << 16));   // tensor_dim0 hi16 | dim1 lo16
  g1[3] = (int)((unsigned)tile0 << 16);                 // dim1 hi16=0 | tile_dim0
  g1[4] = 64;                                           // tile_dim1=64 | tile_dim2=0
  g1[5] = ldw;                                          // tensor_dim0_stride lo32
  g1[6] = 0;                                            // stride hi16 | dim1_stride lo16
  g1[7] = 0;
  i32x4 z4 = {0, 0, 0, 0};
#if defined(__clang_major__) && (__clang_major__ >= 23)
  i32x8 z8 = {0, 0, 0, 0, 0, 0, 0, 0};
  __builtin_amdgcn_tensor_load_to_lds(g0, g1, z4, z4, z8, 0);
#else
  __builtin_amdgcn_tensor_load_to_lds(g0, g1, z4, z4, 0);
#endif
}
#endif

// ---------- weight packing (f32 row-major -> bf16 col-major [N][Ktot]) ----------
__global__ void k_pack_wcat(const float* __restrict__ Kw, const float* __restrict__ Rw,
                            int K0, int Ktot, int N, u16* __restrict__ WT) {
  int idx = blockIdx.x * blockDim.x + threadIdx.x;
  int total = N * Ktot;
  if (idx >= total) return;
  int n = idx / Ktot, k = idx - n * Ktot;
  float v = (k < K0) ? Kw[k * N + n] : Rw[(k - K0) * N + n];
  WT[idx] = f2bf(v);
}

__global__ void k_pack_wp(const float* __restrict__ W, u16* __restrict__ WT) {
  int idx = blockIdx.x * blockDim.x + threadIdx.x;   // 64*512
  if (idx >= D_ * H_) return;
  int n = idx / H_, k = idx - n * H_;
  WT[idx] = f2bf(W[k * D_ + n]);
}

// t=0 input -> bf16 (no imputation at t=0)
__global__ void k_x0(const float* __restrict__ input, u16* __restrict__ curx) {
  int idx = blockIdx.x * blockDim.x + threadIdx.x;   // B*D
  if (idx >= B_ * D_) return;
  int b = idx / D_, d = idx - b * D_;
  curx[idx] = f2bf(input[(size_t)b * T_ * D_ + d]);
}

// ---------- fused LSTM pre-activation GEMM with TDM-staged weights ----------
// Z = [A0|A1] @ WT^T.  Block = 8 waves sharing one 64-column N-strip; the strip's
// K-chunks are DMA'd into double-buffered LDS by the Tensor Data Mover and read
// back as B fragments via ds_load_b128 with immediate offsets (TILE0 constexpr).
// Requirement: nkc0 % KCSUB == 0 so every chunk is sourced from a single A matrix.
template <int KCSUB>
__global__ void __launch_bounds__(256)
k_gemm_dual(const u16* __restrict__ A0, int lda0, int nkc0,
            const u16* __restrict__ A1, int lda1, int nkc1,
            const u16* __restrict__ WT, int ldw,
            float* __restrict__ Z, int N) {
  constexpr int TILE0 = KCSUB * 32;            // K elems per staged chunk
  constexpr int CHUNK = 64 * TILE0;            // elems per LDS buffer
  extern __shared__ u16 smem[];                // [2][64][TILE0] bf16
  int lane = threadIdx.x & 31;
  int wv   = threadIdx.x >> 5;                 // 0..7
  int nstrips = N >> 6;
  int ns  = blockIdx.x % nstrips;
  int mtg = blockIdx.x / nstrips;
  int mt  = mtg * 8 + wv;
  int l15 = lane & 15;
  int hf  = lane >> 4;
  int row = mt * 16 + l15;

  int nchunk = (nkc0 + nkc1) / KCSUB;

  v8f acc[4];
#pragma unroll
  for (int s = 0; s < 4; ++s)
#pragma unroll
    for (int i = 0; i < 8; ++i) acc[s][i] = 0.0f;

  const u16* a0p = A0 + (size_t)row * lda0 + hf * 16;
  const u16* a1p = A1 + (size_t)row * lda1 + hf * 16;

#if HAVE_TDM
  unsigned ldsBase = __builtin_amdgcn_groupstaticsize();
  if (wv == 0)                                  // prefetch chunk 0
    tdm_load_strip(WT, ldw, ns * 64, 0, TILE0, ldsBase);
#endif

  for (int c = 0; c < nchunk; ++c) {
#if HAVE_TDM
    if (wv == 0) __builtin_amdgcn_s_wait_tensorcnt(0);   // chunk c resident
    __syncthreads();                                     // publish to all waves
    if (wv == 0 && c + 1 < nchunk)                       // prefetch chunk c+1
      tdm_load_strip(WT, ldw, ns * 64, (c + 1) * TILE0, TILE0,
                     ldsBase + (unsigned)(((c + 1) & 1) * CHUNK * 2));
#else
    // fallback: cooperative global->LDS copy of chunk c
    __syncthreads();
    {
      constexpr int VROW = TILE0 / 8;                    // 16B vectors per row
      for (int e = threadIdx.x; e < CHUNK / 8; e += 256) {
        int r = e / VROW, k8 = e - r * VROW;
        *(v4u*)(smem + (size_t)(c & 1) * CHUNK + r * TILE0 + k8 * 8) =
            *(const v4u*)(WT + (size_t)(ns * 64 + r) * ldw + c * TILE0 + k8 * 8);
      }
    }
    __syncthreads();
#endif
    // Per-chunk uniform A source (chunk never straddles the A0/A1 boundary).
    int kglob = c * KCSUB;
    const u16* ac = (kglob < nkc0) ? (a0p + (size_t)kglob * 32)
                                   : (a1p + (size_t)(kglob - nkc0) * 32);
    // Per-lane LDS base; all fragment offsets below are compile-time immediates.
    const u16* bbase = smem + (size_t)(c & 1) * CHUNK + (size_t)l15 * TILE0 + hf * 16;
#pragma unroll
    for (int kk = 0; kk < KCSUB; ++kk) {
      v16bf a  = loadA(ac + kk * 32);
      v16bf bf0 = loadB(bbase + kk * 32);
      v16bf bf1 = loadB(bbase + 16 * TILE0 + kk * 32);
      v16bf bf2 = loadB(bbase + 32 * TILE0 + kk * 32);
      v16bf bf3 = loadB(bbase + 48 * TILE0 + kk * 32);
      acc[0] = wmma_bf16(a, bf0, acc[0]);
      acc[1] = wmma_bf16(a, bf1, acc[1]);
      acc[2] = wmma_bf16(a, bf2, acc[2]);
      acc[3] = wmma_bf16(a, bf3, acc[3]);
    }
    __syncthreads();                                     // done reading buf(c)
  }

  // D layout: vgpr v -> M = v + 8*(lane>>4); lane&15 -> N
  int c0 = ns * 64 + l15;
#pragma unroll
  for (int s = 0; s < 4; ++s)
#pragma unroll
    for (int v = 0; v < 8; ++v)
      Z[(size_t)(mt * 16 + hf * 8 + v) * N + c0 + s * 16] = acc[s][v];
}

// ---------- imputation GEMM: pred = h1_prev @ W + pb; also emits prediction rows ----------
__global__ void __launch_bounds__(256)
k_impute(const u16* __restrict__ h1bf, const u16* __restrict__ WpT,
         const float* __restrict__ pb, const float* __restrict__ input,
         int t, u16* __restrict__ curx, float* __restrict__ predOut) {
  int lane = threadIdx.x & 31;
  int mt = (blockIdx.x * blockDim.x + threadIdx.x) >> 5;   // 0..31
  int l15 = lane & 15;
  int hf  = lane >> 4;
  int row = mt * 16 + l15;

  v8f acc[4];
#pragma unroll
  for (int s = 0; s < 4; ++s)
#pragma unroll
    for (int i = 0; i < 8; ++i) acc[s][i] = 0.0f;

  const u16* ap  = h1bf + (size_t)row * H_ + hf * 16;
  const u16* bp0 = WpT + (size_t)l15 * H_ + hf * 16;

#pragma unroll 4
  for (int kc = 0; kc < H_ / 32; ++kc) {
    v16bf a = loadA(ap + kc * 32);
    const u16* bp = bp0 + kc * 32;
    v16bf bf0 = loadB(bp);
    v16bf bf1 = loadB(bp + 16 * H_);
    v16bf bf2 = loadB(bp + 32 * H_);
    v16bf bf3 = loadB(bp + 48 * H_);
    acc[0] = wmma_bf16(a, bf0, acc[0]);
    acc[1] = wmma_bf16(a, bf1, acc[1]);
    acc[2] = wmma_bf16(a, bf2, acc[2]);
    acc[3] = wmma_bf16(a, bf3, acc[3]);
  }
#pragma unroll
  for (int s = 0; s < 4; ++s) {
    int d = s * 16 + l15;
    float pbv = pb[d];
#pragma unroll
    for (int v = 0; v < 8; ++v) {
      int bb = mt * 16 + hf * 8 + v;
      float pred = acc[s][v] + pbv;
      float xv = input[((size_t)bb * T_ + t) * D_ + d];
      float cur = (xv == MISSING_V) ? pred : xv;
      curx[bb * D_ + d] = f2bf(cur);
      predOut[((size_t)bb * (T_ - 1) + (t - 1)) * D_ + d] = pred;
    }
  }
}

// ---------- LSTM gates: z -> (h, c), also bf16 copy of h ----------
__global__ void k_gates(const float* __restrict__ Z, const float* __restrict__ bias,
                        float* __restrict__ h, float* __restrict__ c,
                        u16* __restrict__ hbf) {
  int idx = blockIdx.x * blockDim.x + threadIdx.x;   // B*H
  if (idx >= B_ * H_) return;
  int b = idx >> 9, j = idx & (H_ - 1);
  const float* z = Z + (size_t)b * N4H;
  float zi = z[j]            + bias[j];
  float zf = z[H_ + j]       + bias[H_ + j];
  float zg = z[2 * H_ + j]   + bias[2 * H_ + j];
  float zo = z[3 * H_ + j]   + bias[3 * H_ + j];
  float cn = sigm(zf) * c[idx] + sigm(zi) * tanhf(zg);
  float hn = sigm(zo) * tanhf(cn);
  c[idx] = cn;
  h[idx] = hn;
  hbf[idx] = f2bf(hn);
}

__global__ void k_copy(const float* __restrict__ src, float* __restrict__ dst, int n) {
  int idx = blockIdx.x * blockDim.x + threadIdx.x;
  if (idx < n) dst[idx] = src[idx];
}

// ---------- masked MSE reduction over prediction ----------
__global__ void k_losspred(const float* __restrict__ pred, const float* __restrict__ pt,
                           const float* __restrict__ m, float* __restrict__ acc) {
  __shared__ float red[256];
  float s = 0.0f;
  for (long long i = (long long)blockIdx.x * 256 + threadIdx.x; i < NPRED;
       i += (long long)gridDim.x * 256) {
    float v = (pt[i] - pred[i]) * m[i];
    s += v * v;
  }
  red[threadIdx.x] = s; __syncthreads();
  for (int k = 128; k > 0; k >>= 1) {
    if ((int)threadIdx.x < k) red[threadIdx.x] += red[threadIdx.x + k];
    __syncthreads();
  }
  if (threadIdx.x == 0) atomicAdd(acc, red[0]);
}

// ---------- sum of squares (L2 reg) ----------
__global__ void k_sumsq(const float* __restrict__ p, int n, float* __restrict__ acc) {
  __shared__ float red[256];
  float s = 0.0f;
  for (int i = blockIdx.x * 256 + threadIdx.x; i < n; i += gridDim.x * 256) {
    float v = p[i];
    s += v * v;
  }
  red[threadIdx.x] = s; __syncthreads();
  for (int k = 128; k > 0; k >>= 1) {
    if ((int)threadIdx.x < k) red[threadIdx.x] += red[threadIdx.x + k];
    __syncthreads();
  }
  if (threadIdx.x == 0) atomicAdd(acc, red[0]);
}

__global__ void k_finalize(const float* __restrict__ lp_acc, const float* __restrict__ reg_acc,
                           float* __restrict__ lp_final, float* __restrict__ out_lp,
                           float* __restrict__ out_reg) {
  float lp = lp_acc[0] / ((float)NPRED * (float)B_);   // mean then /B
  lp_final[0] = lp;
  out_lp[0]   = lp;
  out_reg[0]  = 0.5f * reg_acc[0];
}

// ---------- classification head + per-batch losses ----------
__global__ void k_classify(const float* __restrict__ h1, const float* __restrict__ dw,
                           const float* __restrict__ db, const float* __restrict__ ltgt,
                           const float* __restrict__ lp_final,
                           float* __restrict__ out_loss, float* __restrict__ out_acc,
                           float* __restrict__ out_lbl) {
  __shared__ float red[256];
  __shared__ float logits[C_];
  int b = blockIdx.x, tid = threadIdx.x;
  float part[C_];
  for (int c = 0; c < C_; ++c) part[c] = 0.0f;
  for (int k = tid; k < H_; k += 256) {
    float hv = h1[(size_t)b * H_ + k];
    for (int c = 0; c < C_; ++c) part[c] += hv * dw[k * C_ + c];
  }
  for (int c = 0; c < C_; ++c) {
    red[tid] = part[c]; __syncthreads();
    for (int s = 128; s > 0; s >>= 1) {
      if (tid < s) red[tid] += red[tid + s];
      __syncthreads();
    }
    if (tid == 0) logits[c] = red[0] + db[c];
    __syncthreads();
  }
  if (tid == 0) {
    float mx = logits[0];
    for (int c = 1; c < C_; ++c) mx = fmaxf(mx, logits[c]);
    float ex[C_], se = 0.0f;
    for (int c = 0; c < C_; ++c) { ex[c] = expf(logits[c] - mx); se += ex[c]; }
    float lse = logf(se);
    float closs = 0.0f, bp = -1e30f, bt = -1e30f;
    int am_p = 0, am_t = 0;
    for (int c = 0; c < C_; ++c) {
      float sm = ex[c] / se;
      out_lbl[b * C_ + c] = sm;
      closs -= ltgt[b * C_ + c] * (logits[c] - mx - lse);
      if (sm > bp) { bp = sm; am_p = c; }
      float lt = ltgt[b * C_ + c];
      if (lt > bt) { bt = lt; am_t = c; }
    }
    out_loss[b] = closs + 1.0f * lp_final[0];   // LAMDA = 1.0
    out_acc[b]  = (am_p == am_t) ? 1.0f : 0.0f;
  }
}

// ---------- host orchestration ----------
extern "C" void kernel_launch(void* const* d_in, const int* in_sizes, int n_in,
                              void* d_out, int out_size, void* d_ws, size_t ws_size,
                              hipStream_t stream) {
  (void)in_sizes; (void)n_in; (void)out_size;
  const float* input = (const float*)d_in[0];
  const float* pt    = (const float*)d_in[1];
  const float* mask  = (const float*)d_in[2];
  const float* ltgt  = (const float*)d_in[3];
  const float* k0    = (const float*)d_in[4];
  const float* r0    = (const float*)d_in[5];
  const float* b0    = (const float*)d_in[6];
  const float* k1    = (const float*)d_in[7];
  const float* r1    = (const float*)d_in[8];
  const float* b1    = (const float*)d_in[9];
  const float* W     = (const float*)d_in[10];
  const float* pb    = (const float*)d_in[11];
  const float* dw    = (const float*)d_in[12];
  const float* db    = (const float*)d_in[13];
  float* out = (float*)d_out;

  // output layout (floats)
  const size_t LOSS_OFF = 0;
  const size_t ACC_OFF  = 512;
  const size_t PRED_OFF = 1024;
  const size_t LBL_OFF  = PRED_OFF + (size_t)NPRED;
  const size_t LH_OFF   = LBL_OFF + (size_t)B_ * C_;
  const size_t LP_OFF   = LH_OFF + (size_t)B_ * H_;
  const size_t REG_OFF  = LP_OFF + 1;

  // workspace layout (bytes)
  char* ws = (char*)d_ws;
  const size_t OFF_WT0  = 0;                               // bf16 [2048][576]
  const size_t OFF_WT1  = OFF_WT0 + (size_t)N4H * 576 * 2; // bf16 [2048][1024]
  const size_t OFF_WPT  = OFF_WT1 + (size_t)N4H * 1024 * 2;// bf16 [64][512]
  const size_t OFF_CURX = OFF_WPT + (size_t)D_ * H_ * 2;   // bf16 [B][64]
  const size_t OFF_Z    = OFF_CURX + (size_t)B_ * D_ * 2;  // f32 [B][2048]
  const size_t OFF_ZERO = OFF_Z + (size_t)B_ * N4H * 4;    // zero-init region:
  const size_t OFF_H0   = OFF_ZERO;
  const size_t OFF_C0   = OFF_H0 + (size_t)B_ * H_ * 4;
  const size_t OFF_H1   = OFF_C0 + (size_t)B_ * H_ * 4;
  const size_t OFF_C1   = OFF_H1 + (size_t)B_ * H_ * 4;
  const size_t OFF_H0BF = OFF_C1 + (size_t)B_ * H_ * 4;
  const size_t OFF_H1BF = OFF_H0BF + (size_t)B_ * H_ * 2;
  const size_t OFF_ACC  = OFF_H1BF + (size_t)B_ * H_ * 2;
  const size_t ZERO_BYTES = (OFF_ACC + 256) - OFF_ZERO;
  if (ws_size < OFF_ACC + 256) return;

  u16*   WT0   = (u16*)(ws + OFF_WT0);
  u16*   WT1   = (u16*)(ws + OFF_WT1);
  u16*   WPT   = (u16*)(ws + OFF_WPT);
  u16*   curx  = (u16*)(ws + OFF_CURX);
  float* Z     = (float*)(ws + OFF_Z);
  float* h0    = (float*)(ws + OFF_H0);
  float* c0    = (float*)(ws + OFF_C0);
  float* h1    = (float*)(ws + OFF_H1);
  float* c1    = (float*)(ws + OFF_C1);
  u16*   h0bf  = (u16*)(ws + OFF_H0BF);
  u16*   h1bf  = (u16*)(ws + OFF_H1BF);
  float* lp_acc   = (float*)(ws + OFF_ACC);
  float* reg_acc  = lp_acc + 1;
  float* lp_final = lp_acc + 2;

  // zero states + accumulators every call (ws is not re-poisoned between replays)
  hipMemsetAsync(ws + OFF_ZERO, 0, ZERO_BYTES, stream);

  // pack weights to bf16, pre-transposed for contiguous B-fragment loads
  k_pack_wcat<<<(N4H * 576 + 255) / 256, 256, 0, stream>>>(k0, r0, D_, 576, N4H, WT0);
  k_pack_wcat<<<(N4H * 1024 + 255) / 256, 256, 0, stream>>>(k1, r1, H_, 1024, N4H, WT1);
  k_pack_wp<<<(D_ * H_ + 255) / 256, 256, 0, stream>>>(W, WPT);

  // t = 0 input (no imputation)
  k_x0<<<(B_ * D_ + 255) / 256, 256, 0, stream>>>(input, curx);

  // dynamic LDS: double-buffered [2][64][KCSUB*32] bf16 strips
  // layer0: KCSUB=2 -> chunk 0 entirely curx (nkc0=2), chunks 1..8 h0bf; 16 KB LDS
  // layer1: KCSUB=4 -> chunks 0..3 h0bf, 4..7 h1bf; 32 KB LDS
  const size_t SMEM0 = 2ull * 64 * (2 * 32) * 2;
  const size_t SMEM1 = 2ull * 64 * (4 * 32) * 2;

  // recurrent time loop
  for (int t = 0; t < T_; ++t) {
    if (t > 0) {
      k_impute<<<4, 256, 0, stream>>>(h1bf, WPT, pb, input, t, curx, out + PRED_OFF);
    }
    // layer 0: [curx(64) | h0(512)] @ [k0;r0]
    k_gemm_dual<2><<<128, 256, SMEM0, stream>>>(curx, D_, 2, h0bf, H_, 16,
                                                WT0, 576, Z, N4H);
    k_gates<<<(B_ * H_) / 256, 256, 0, stream>>>(Z, b0, h0, c0, h0bf);
    // layer 1: [h0(512) | h1(512)] @ [k1;r1]
    k_gemm_dual<4><<<128, 256, SMEM1, stream>>>(h0bf, H_, 16, h1bf, H_, 16,
                                                WT1, 1024, Z, N4H);
    k_gates<<<(B_ * H_) / 256, 256, 0, stream>>>(Z, b1, h1, c1, h1bf);
  }

  // last hidden state
  k_copy<<<(B_ * H_) / 256, 256, 0, stream>>>(h1, out + LH_OFF, B_ * H_);

  // losses + reg
  k_losspred<<<2048, 256, 0, stream>>>(out + PRED_OFF, pt, mask, lp_acc);
  k_sumsq<<<512, 256, 0, stream>>>(k0, D_ * N4H, reg_acc);
  k_sumsq<<<1024, 256, 0, stream>>>(r0, H_ * N4H, reg_acc);
  k_sumsq<<<8, 256, 0, stream>>>(b0, N4H, reg_acc);
  k_sumsq<<<1024, 256, 0, stream>>>(k1, H_ * N4H, reg_acc);
  k_sumsq<<<1024, 256, 0, stream>>>(r1, H_ * N4H, reg_acc);
  k_sumsq<<<8, 256, 0, stream>>>(b1, N4H, reg_acc);
  k_sumsq<<<128, 256, 0, stream>>>(W, H_ * D_, reg_acc);
  k_sumsq<<<1, 256, 0, stream>>>(pb, D_, reg_acc);
  k_sumsq<<<20, 256, 0, stream>>>(dw, H_ * C_, reg_acc);
  k_sumsq<<<1, 256, 0, stream>>>(db, C_, reg_acc);

  k_finalize<<<1, 1, 0, stream>>>(lp_acc, reg_acc, lp_final,
                                  out + LP_OFF, out + REG_OFF);

  k_classify<<<B_, 256, 0, stream>>>(h1, dw, db, ltgt, lp_final,
                                     out + LOSS_OFF, out + ACC_OFF, out + LBL_OFF);
}
